// SelfAttention_29540785062594
// MI455X (gfx1250) — matrix operand order repaired and checked by reading, not compiled, not run on hardware
//
#include <hip/hip_runtime.h>
#include <stdint.h>

typedef __bf16 bf16_t;
typedef bf16_t v16bf __attribute__((ext_vector_type(16)));
typedef bf16_t v8bf  __attribute__((ext_vector_type(8)));
typedef float  v8f   __attribute__((ext_vector_type(8)));
typedef uint32_t u32x4 __attribute__((ext_vector_type(4)));
typedef int      i32x4 __attribute__((ext_vector_type(4)));
typedef int      i32x8 __attribute__((ext_vector_type(8)));

// ---------------------------------------------------------------------------
// Problem constants (reference: B=8, H=W=64, FIN=256, INNER=32)
// ---------------------------------------------------------------------------
static constexpr int NB    = 8;
static constexpr int NPIX  = 4096;   // 64*64
static constexpr int NPOOL = 1024;   // 32*32

// ---------------------------------------------------------------------------
// Tensor Data Mover: 2-D tile DMA  global -> LDS  (CDNA5 ISA ch. 7/8)
// D# group0: [1:0]=count, [63:32]=lds_addr, [120:64]=global_addr, [127:126]=2
// D# group1: [17:16]=data_size, [79:48]=tensor_dim0, [111:80]=tensor_dim1,
//            [127:112]=tile_dim0, [143:128]=tile_dim1, [207:160]=dim0_stride
// ---------------------------------------------------------------------------
__device__ __forceinline__ void tdm_load_to_lds_2d(
    uint32_t lds_off, const void* gptr, uint32_t dsize_code,
    uint32_t tensor_d0, uint32_t tensor_d1,
    uint32_t tile_d0, uint32_t tile_d1, uint32_t stride0) {
#if __has_builtin(__builtin_amdgcn_tensor_load_to_lds)
  uint64_t ga = (uint64_t)gptr;
  u32x4 g0;
  g0[0] = 1u;                                                   // count=1
  g0[1] = lds_off;                                              // lds_addr
  g0[2] = (uint32_t)ga;                                         // addr[31:0]
  g0[3] = (uint32_t)((ga >> 32) & 0x01FFFFFFu) | (2u << 30);    // addr[56:32]|type=2
  i32x8 g1;
  g1[0] = (int)(dsize_code << 16);                              // data_size
  g1[1] = (int)((tensor_d0 & 0xFFFFu) << 16);                   // tensor_dim0 lo
  g1[2] = (int)(((tensor_d0 >> 16) & 0xFFFFu) | ((tensor_d1 & 0xFFFFu) << 16));
  g1[3] = (int)(((tensor_d1 >> 16) & 0xFFFFu) | ((tile_d0 & 0xFFFFu) << 16));
  g1[4] = (int)(tile_d1 & 0xFFFFu);                             // tile_dim1, tile_dim2=0
  g1[5] = (int)stride0;                                         // dim0_stride lo
  g1[6] = 0;
  g1[7] = 0;
  i32x4 z4 = {0, 0, 0, 0};
#if defined(__clang_major__) && (__clang_major__ >= 23)
  i32x8 z8 = {0, 0, 0, 0, 0, 0, 0, 0};
  __builtin_amdgcn_tensor_load_to_lds(g0, g1, z4, z4, z8, 0);
#else
  __builtin_amdgcn_tensor_load_to_lds(g0, g1, z4, z4, 0);
#endif
#endif
}

// ---------------------------------------------------------------------------
// WMMA fragment helpers: layouts per CDNA5 ISA 7.12.2
//   A 16x32 bf16: lane<16 -> row=lane,  K chunks {0..7} and {16..23}
//                 lane>=16-> row=lane-16, K chunks {8..15} and {24..31}
//   B 32x16 bf16: lane<16 -> col=lane,  K {0..7},{16..23} ; lane>=16 -> col=lane-16, K {8..15},{24..31}
//   C 16x16 f32 : elem i of v8f at lane -> M = i + 8*(lane>=16), N = lane%16
// ---------------------------------------------------------------------------
__device__ __forceinline__ v8f wmma_bf16(v16bf a, v16bf b, v8f c) {
  return __builtin_amdgcn_wmma_f32_16x16x32_bf16(
      false, a, false, b, (short)0, c, false, false);
}

__device__ __forceinline__ v16bf loadA(const bf16_t* __restrict__ base, int ld, int k0) {
  const int lane = threadIdx.x & 31;
  const int half = lane >> 4;
  const int r    = lane & 15;
  const bf16_t* p = base + r * ld + k0 + (half ? 8 : 0);
  v8bf lo = *(const v8bf*)p;
  v8bf hi = *(const v8bf*)(p + 16);
  v16bf a;
#pragma unroll
  for (int i = 0; i < 8; ++i) { a[i] = lo[i]; a[i + 8] = hi[i]; }
  return a;
}

// A fragment straight from an fp32 tile in LDS (convert at use site)
__device__ __forceinline__ v16bf loadA_f32(const float* __restrict__ base, int ld, int k0) {
  const int lane = threadIdx.x & 31;
  const int half = lane >> 4;
  const int r    = lane & 15;
  const float* p = base + r * ld + k0 + (half ? 8 : 0);
  float4 a0 = *(const float4*)(p);
  float4 a1 = *(const float4*)(p + 4);
  float4 b0 = *(const float4*)(p + 16);
  float4 b1 = *(const float4*)(p + 20);
  v16bf a;
  a[0]  = (bf16_t)a0.x; a[1]  = (bf16_t)a0.y; a[2]  = (bf16_t)a0.z; a[3]  = (bf16_t)a0.w;
  a[4]  = (bf16_t)a1.x; a[5]  = (bf16_t)a1.y; a[6]  = (bf16_t)a1.z; a[7]  = (bf16_t)a1.w;
  a[8]  = (bf16_t)b0.x; a[9]  = (bf16_t)b0.y; a[10] = (bf16_t)b0.z; a[11] = (bf16_t)b0.w;
  a[12] = (bf16_t)b1.x; a[13] = (bf16_t)b1.y; a[14] = (bf16_t)b1.z; a[15] = (bf16_t)b1.w;
  return a;
}

__device__ __forceinline__ v16bf loadB(const bf16_t* __restrict__ base, int ld, int k0) {
  const int lane = threadIdx.x & 31;
  const int half = lane >> 4;
  const int r    = lane & 15;
  const bf16_t* p = base + r * ld + k0 + (half ? 8 : 0);
  v8bf lo = *(const v8bf*)p;
  v8bf hi = *(const v8bf*)(p + 16);
  v16bf b;
#pragma unroll
  for (int i = 0; i < 8; ++i) { b[i] = lo[i]; b[i + 8] = hi[i]; }
  return b;
}

// ---------------------------------------------------------------------------
// Kernel 1: repack weights to bf16 in WMMA-B-friendly (N-major) layout.
// ---------------------------------------------------------------------------
__global__ __launch_bounds__(256) void prep_kernel(
    const float* __restrict__ Wf, const float* __restrict__ Wg,
    const float* __restrict__ Wh, const float* __restrict__ Wattn,
    bf16_t* __restrict__ WpackT, bf16_t* __restrict__ WattnT) {
  int t = blockIdx.x * 256 + threadIdx.x;          // 0 .. 32767
  if (t < 96 * 256) {
    int n = t >> 8, k = t & 255;
    float v = (n < 32) ? Wf[k * 32 + n]
            : (n < 64) ? Wg[k * 32 + (n - 32)]
                       : Wh[k * 32 + (n - 64)];
    WpackT[n * 256 + k] = (bf16_t)v;
  } else {
    int u = t - 96 * 256;
    int c = u >> 5, d = u & 31;
    WattnT[c * 32 + d] = (bf16_t)Wattn[d * 256 + c];
  }
}

// ---------------------------------------------------------------------------
// Kernel 2: fused projection  fgh = x @ [Wf|Wg|Wh]  (32768 x 256)*(256 x 96)
//   x tile (128 x 256 fp32 = 128 KB) DMA'd into LDS by the TDM; converted to
//   bf16 at A-fragment load time (amortized over 6 WMMAs per fragment).
// ---------------------------------------------------------------------------
__global__ __launch_bounds__(256) void proj_kernel(
    const float* __restrict__ x, const bf16_t* __restrict__ WpackT,
    bf16_t* __restrict__ fgh) {
  __shared__ __align__(16) char smemP[128 * 256 * 4];   // 128 KB fp32 tile
  float* sX = (float*)smemP;
  const int row0 = blockIdx.x * 128;

#if __has_builtin(__builtin_amdgcn_tensor_load_to_lds)
  if ((threadIdx.x >> 5) == 0) {
    // 2-D tile: 256 fp32 per row x 128 rows, row stride 256 elements
    tdm_load_to_lds_2d(0u, (const void*)(x + (size_t)row0 * 256),
                       2u /*4B*/, 256u, 128u, 256u, 128u, 256u);
    __builtin_amdgcn_s_wait_tensorcnt(0);
  }
  __syncthreads();
#else
  for (int i = threadIdx.x; i < 128 * 64; i += 256) {
    ((float4*)sX)[i] = ((const float4*)x)[(size_t)row0 * 64 + i];
  }
  __syncthreads();
#endif

  const int w = threadIdx.x >> 5;
  v8f acc[6] = {};
  for (int k0 = 0; k0 < 256; k0 += 32) {
    v16bf a = loadA_f32(sX + w * 16 * 256, 256, k0);
#pragma unroll
    for (int ni = 0; ni < 6; ++ni) {
      v16bf b = loadB(WpackT + ni * 16 * 256, 256, k0);
      acc[ni] = wmma_bf16(a, b, acc[ni]);
    }
  }

  const int lane = threadIdx.x & 31, half = lane >> 4, r = lane & 15;
#pragma unroll
  for (int ni = 0; ni < 6; ++ni) {
    bf16_t* dst = fgh + (size_t)(row0 + w * 16 + 8 * half) * 96 + ni * 16 + r;
#pragma unroll
    for (int i = 0; i < 8; ++i) dst[(size_t)i * 96] = (bf16_t)acc[ni][i];
  }
}

// ---------------------------------------------------------------------------
// Kernel 3: 2x2 maxpool of f' and h' columns of fgh.
//   fP : (B, 1024, 32) bf16 key-major  -> B operand of QK^T
//   hT : (B, 32, 1024) bf16 feat-major -> B operand of P*V
// ---------------------------------------------------------------------------
__global__ __launch_bounds__(256) void pool_kernel(
    const bf16_t* __restrict__ fgh, bf16_t* __restrict__ fP,
    bf16_t* __restrict__ hT) {
  int t = blockIdx.x * 256 + threadIdx.x;
  int d  = t & 31;
  int qp = (t >> 5) & 1023;
  int b  = t >> 15;
  int i = qp >> 5, j = qp & 31;
  int q00 = (2 * i) * 64 + 2 * j;
  float fm = -3.4e38f, hm = -3.4e38f;
#pragma unroll
  for (int di = 0; di < 2; ++di)
#pragma unroll
    for (int dj = 0; dj < 2; ++dj) {
      size_t idx = ((size_t)(b * NPIX + q00 + di * 64 + dj)) * 96;
      fm = fmaxf(fm, (float)fgh[idx + d]);
      hm = fmaxf(hm, (float)fgh[idx + 64 + d]);
    }
  fP[((size_t)(b * NPOOL + qp)) * 32 + d] = (bf16_t)fm;
  hT[((size_t)(b * 32 + d)) * 1024 + qp] = (bf16_t)hm;
}

// ---------------------------------------------------------------------------
// Kernel 4: fused attention + output projection + residual.
//   One block = (batch b, 32-query tile). 8 waves.
//   LDS carve (single block, manual offsets so the TDM lds_addr is known):
//     [0      , 131072) : S scores fp32 32x1024 (reused for O partials)
//     [131072 , 196608) : P probabilities bf16 32x1024
//     [196608 , 262144) : V = hT batch block bf16 32x1024  (TDM-prefetched)
//     [262144 , 264192) : O in bf16 32x32
//   The hT DMA overlaps Phases 1+2 entirely; waited just before Phase 3.
// ---------------------------------------------------------------------------
__global__ __launch_bounds__(256) void attn_kernel(
    const bf16_t* __restrict__ fgh, const bf16_t* __restrict__ fP,
    const bf16_t* __restrict__ hT, const bf16_t* __restrict__ WattnT,
    const float* __restrict__ x, const float* __restrict__ gamma,
    float* __restrict__ out) {
  __shared__ __align__(16) char smem[264192];
  float*  sS   = (float*)(smem);
  bf16_t* sP   = (bf16_t*)(smem + 131072);
  bf16_t* sV   = (bf16_t*)(smem + 196608);
  bf16_t* sObf = (bf16_t*)(smem + 262144);

  const int w    = threadIdx.x >> 5;
  const int lane = threadIdx.x & 31;
  const int half = lane >> 4;
  const int r    = lane & 15;
  const int b    = blockIdx.x >> 7;
  const int q0   = (blockIdx.x & 127) * 32;

  const bf16_t* fb  = fP + (size_t)b * NPOOL * 32;
  const bf16_t* hTb = hT + (size_t)b * 32 * 1024;

  // Kick off async DMA of the whole hT batch block (64 KB) into LDS; it
  // overlaps the QK^T and softmax phases below.
#if __has_builtin(__builtin_amdgcn_tensor_load_to_lds)
  if (w == 0) {
    tdm_load_to_lds_2d(196608u, (const void*)hTb,
                       3u /*8B*/, 8192u, 1u, 8192u, 1u, 8192u);
  }
#else
  for (int i = threadIdx.x; i < 8192; i += 256) {
    ((uint64_t*)sV)[i] = ((const uint64_t*)hTb)[i];
  }
#endif
  // Warm L2/L0 for the f (key) block consumed by Phase 1.
  __builtin_prefetch(fb + (size_t)threadIdx.x * 128, 0, 1);

  // ---- Phase 1: S = g(queries) @ f^T   (2 M-tiles x 64 N-tiles, K=32) ----
  const bf16_t* gbase = fgh + ((size_t)(b * NPIX + q0)) * 96 + 32;  // g cols
#pragma unroll
  for (int mi = 0; mi < 2; ++mi) {
    v16bf a = loadA(gbase + (size_t)(16 * mi) * 96, 96, 0);
#pragma unroll
    for (int j = 0; j < 8; ++j) {
      int ni = w * 8 + j;
      v16bf bf = loadB(fb + (size_t)ni * 16 * 32, 32, 0);
      v8f c = {};
      c = wmma_bf16(a, bf, c);
      float* dst = sS + (size_t)(16 * mi + 8 * half) * 1024 + ni * 16 + r;
#pragma unroll
      for (int i = 0; i < 8; ++i) dst[(size_t)i * 1024] = c[i];
    }
  }
  __syncthreads();

  // ---- Phase 2: row softmax in fp32, P written as bf16 (4 rows / wave) ----
#pragma unroll
  for (int rr = 0; rr < 4; ++rr) {
    int row = w * 4 + rr;
    const float* srow = sS + (size_t)row * 1024;
    float ev[32];
    float m = -3.4e38f;
#pragma unroll
    for (int i = 0; i < 32; ++i) {
      float s = srow[lane + i * 32];
      ev[i] = s;
      m = fmaxf(m, s);
    }
#pragma unroll
    for (int off = 16; off > 0; off >>= 1) m = fmaxf(m, __shfl_xor(m, off, 32));
    float sum = 0.f;
#pragma unroll
    for (int i = 0; i < 32; ++i) { float e = __expf(ev[i] - m); ev[i] = e; sum += e; }
#pragma unroll
    for (int off = 16; off > 0; off >>= 1) sum += __shfl_xor(sum, off, 32);
    float inv = 1.f / sum;
    bf16_t* prow = sP + (size_t)row * 1024;
#pragma unroll
    for (int i = 0; i < 32; ++i) prow[lane + i * 32] = (bf16_t)(ev[i] * inv);
  }
#if __has_builtin(__builtin_amdgcn_tensor_load_to_lds)
  if (w == 0) __builtin_amdgcn_s_wait_tensorcnt(0);   // hT tile now in LDS
#endif
  __syncthreads();

  // ---- Phase 3: O = P @ V  (K=1024 split over 8 waves, 128 per wave) ----
  v8f acc[2][2] = {};
  const int kb = w * 128;
#pragma unroll
  for (int ks = 0; ks < 4; ++ks) {
    int k0 = kb + ks * 32;
    v16bf a0 = loadA(sP,             1024, k0);
    v16bf a1 = loadA(sP + 16 * 1024, 1024, k0);
    v16bf b0 = loadB(sV,             1024, k0);
    v16bf b1 = loadB(sV + 16 * 1024, 1024, k0);
    acc[0][0] = wmma_bf16(a0, b0, acc[0][0]);
    acc[0][1] = wmma_bf16(a0, b1, acc[0][1]);
    acc[1][0] = wmma_bf16(a1, b0, acc[1][0]);
    acc[1][1] = wmma_bf16(a1, b1, acc[1][1]);
  }
  {  // stash per-wave partials in the (now free) sS region
    float* base = sS + (size_t)w * 1024;
#pragma unroll
    for (int mi = 0; mi < 2; ++mi)
#pragma unroll
      for (int ni = 0; ni < 2; ++ni) {
        float* dst = base + (mi * 16 + 8 * half) * 32 + ni * 16 + r;
#pragma unroll
        for (int i = 0; i < 8; ++i) dst[i * 32] = acc[mi][ni][i];
      }
  }
  __syncthreads();

  // ---- Phase 4: reduce partials, convert O to bf16 ----
  for (int e = threadIdx.x; e < 1024; e += 256) {
    float s = 0.f;
#pragma unroll
    for (int ww = 0; ww < 8; ++ww) s += sS[(size_t)ww * 1024 + e];
    sObf[e] = (bf16_t)s;
  }
  __syncthreads();

  // ---- Phase 5: out = gamma * (O @ Wattn) + x   (2 x 16 tiles, 4/wave) ----
  const float g0 = gamma[0];
#pragma unroll
  for (int t = 0; t < 4; ++t) {
    int tile = w * 4 + t;
    int mi = tile >> 4, ni = tile & 15;
    v16bf a  = loadA(sObf + mi * 16 * 32, 32, 0);
    v16bf bw = loadB(WattnT + (size_t)ni * 16 * 32, 32, 0);
    v8f c = {};
    c = wmma_bf16(a, bw, c);
    int row0 = q0 + 16 * mi + 8 * half;
    int col  = ni * 16 + r;
#pragma unroll
    for (int i = 0; i < 8; ++i) {
      size_t idx = ((size_t)(b * NPIX + row0 + i)) * 256 + col;
      out[idx] = g0 * c[i] + x[idx];
    }
  }
}

// ---------------------------------------------------------------------------
// Launcher
// ---------------------------------------------------------------------------
extern "C" void kernel_launch(void* const* d_in, const int* in_sizes, int n_in,
                              void* d_out, int out_size, void* d_ws, size_t ws_size,
                              hipStream_t stream) {
  (void)in_sizes; (void)n_in; (void)out_size; (void)ws_size;
  const float* x     = (const float*)d_in[0];
  const float* Wf    = (const float*)d_in[1];
  const float* Wg    = (const float*)d_in[2];
  const float* Wh    = (const float*)d_in[3];
  const float* Wattn = (const float*)d_in[4];
  const float* gamma = (const float*)d_in[5];
  float* out = (float*)d_out;

  char* ws = (char*)d_ws;
  const size_t FGH_B = (size_t)NB * NPIX * 96 * 2;    // 6 MB  bf16 (f'|g|h')
  const size_t FP_B  = (size_t)NB * NPOOL * 32 * 2;   // 512 KB
  const size_t HT_B  = (size_t)NB * 32 * 1024 * 2;    // 512 KB
  const size_t WPK_B = (size_t)96 * 256 * 2;          // 48 KB
  bf16_t* fgh    = (bf16_t*)(ws);
  bf16_t* fP     = (bf16_t*)(ws + FGH_B);
  bf16_t* hT     = (bf16_t*)(ws + FGH_B + FP_B);
  bf16_t* WpackT = (bf16_t*)(ws + FGH_B + FP_B + HT_B);
  bf16_t* WattnT = (bf16_t*)(ws + FGH_B + FP_B + HT_B + WPK_B);

  prep_kernel<<<128, 256, 0, stream>>>(Wf, Wg, Wh, Wattn, WpackT, WattnT);
  proj_kernel<<<(NB * NPIX) / 128, 256, 0, stream>>>(x, WpackT, fgh);
  pool_kernel<<<(NB * NPOOL * 32) / 256, 256, 0, stream>>>(fgh, fP, hT);
  attn_kernel<<<NB * (NPIX / 32), 256, 0, stream>>>(fgh, fP, hT, WattnT, x, gamma, out);
}
